// BLSTM_4922032521713
// MI455X (gfx1250) — compile-verified, matrix operand order checked by reading, not run on hardware
//
#include <hip/hip_runtime.h>

// ---------------------------------------------------------------------------
// BLSTM language model head for MI455X (gfx1250, wave32, WMMA).
// Stages: embed-gather -> bf16 converts -> input-gate GEMM -> persistent
// recurrent scan (LDS state, wmma bf16) -> vocab FC GEMM (wmma bf16 with
// async-to-LDS double-buffered weight staging).
// ---------------------------------------------------------------------------

typedef __attribute__((ext_vector_type(8)))  __bf16 v8bf;
typedef __attribute__((ext_vector_type(16))) __bf16 v16bf;
typedef __attribute__((ext_vector_type(8)))  float  v8f;

#define HDIM 512
#define BDIM 16
#define TDIM 512
#define VDIM 32000
#define G4   2048           // 4*H
#define ROWS 8192           // T*B
#define K2   1024           // 2*H

// ---- WMMA helpers ---------------------------------------------------------
// A tile 16x32 (MxK) bf16: lane L and L+16 hold row (L&15); lanes 0-15 carry
// K = {k0..k0+7, k0+16..k0+23}, lanes 16-31 carry K = {k0+8..15, k0+24..31}.
// Two 16-byte loads per lane. B tile (KxN) uses the identical pattern with
// lane = output column, so [out,in]-row-major weights load with the same code.
__device__ __forceinline__ v16bf load_tile16(const __bf16* __restrict__ base,
                                             int stride, int lane) {
  const int r = lane & 15;
  const int half = lane >> 4;
  const __bf16* p = base + (size_t)r * stride + half * 8;
  v8bf lo = *(const v8bf*)(p);
  v8bf hi = *(const v8bf*)(p + 16);
  v16bf out;
#pragma unroll
  for (int i = 0; i < 8; ++i) { out[i] = lo[i]; out[i + 8] = hi[i]; }
  return out;
}

__device__ __forceinline__ v8f wmma_bf16(v16bf a, v16bf b, v8f c) {
  // (neg_a, A, neg_b, B, c_mod, C, reuse_a, reuse_b)
  return __builtin_amdgcn_wmma_f32_16x16x32_bf16(false, a, false, b,
                                                 (short)0, c, false, false);
}

// Async DMA: global (GV, 64-bit vaddr) -> LDS, 16 bytes/lane, ASYNCcnt-tracked.
__device__ __forceinline__ void async_copy_b128(unsigned int lds_byte,
                                                const void* gptr) {
  asm volatile("global_load_async_to_lds_b128 %0, %1, off"
               :: "v"(lds_byte), "v"((unsigned long long)(uintptr_t)gptr)
               : "memory");
}

__device__ __forceinline__ float sigmoid_(float x) {
  return 1.0f / (1.0f + __expf(-x));
}
__device__ __forceinline__ float tanh_(float x) {
  x = fminf(15.0f, fmaxf(-15.0f, x));
  float e = __expf(-2.0f * x);
  return (1.0f - e) / (1.0f + e);
}

// ---- Stage 1: embedding gather, fp32 -> bf16, time-major rows r = t*16+b ---
__global__ void k_gather_embed(const int* __restrict__ x,
                               const float* __restrict__ emb,
                               __bf16* __restrict__ e) {
  size_t idx = (size_t)blockIdx.x * blockDim.x + threadIdx.x;
  size_t total = (size_t)ROWS * HDIM / 8;
  if (idx >= total) return;
  int k0  = (int)((idx & 63) * 8);   // HDIM/8 = 64 chunks per row
  int row = (int)(idx >> 6);         // row = t*16 + b
  int b = row & 15, t = row >> 4;
  const float* src = emb + (size_t)x[b * TDIM + t] * HDIM + k0;
  v8bf o;
#pragma unroll
  for (int i = 0; i < 8; ++i) o[i] = (__bf16)src[i];
  *(v8bf*)(e + (size_t)row * HDIM + k0) = o;
}

// ---- Stage 2: fp32 -> bf16 weight conversion (8 elems/thread) -------------
__global__ void k_cvt_bf16(const float* __restrict__ s, __bf16* __restrict__ d,
                           long n8) {
  long i = (long)blockIdx.x * blockDim.x + threadIdx.x;
  if (i >= n8) return;
  const float* p = s + i * 8;
  v8bf o;
#pragma unroll
  for (int j = 0; j < 8; ++j) o[j] = (__bf16)p[j];
  *(v8bf*)(d + i * 8) = o;
}

// ---- Stage 3: xg = e @ Wx^T + bx for both directions ----------------------
// One 16x16 output tile per wave; K = 512 -> 16 wmma per tile. L2-resident.
__global__ __launch_bounds__(256) void k_xg_gemm(
    const __bf16* __restrict__ e,
    const __bf16* __restrict__ wfx, const __bf16* __restrict__ wbx,
    const float* __restrict__ bfx, const float* __restrict__ bbx,
    float* __restrict__ xgf, float* __restrict__ xgb) {
  const int lane = threadIdx.x & 31;
  const int wave = threadIdx.x >> 5;
  const int tile = blockIdx.x * 8 + wave;   // 0..65535
  const int mt = tile >> 7;                 // 0..511 (rows of T*B)
  const int nt = tile & 127;                // 0..127 (4H tiles)
  const __bf16* wx = blockIdx.z ? wbx : wfx;
  const float*  bx = blockIdx.z ? bbx : bfx;
  float* xg = blockIdx.z ? xgb : xgf;

  const __bf16* abase = e  + (size_t)mt * 16 * HDIM;
  const __bf16* bbase = wx + (size_t)nt * 16 * HDIM;
  v8f acc = {};
  for (int k0 = 0; k0 < HDIM; k0 += 32) {
    v16bf a = load_tile16(abase + k0, HDIM, lane);
    v16bf b = load_tile16(bbase + k0, HDIM, lane);
    acc = wmma_bf16(a, b, acc);
  }
  const int r = lane & 15, half = lane >> 4;
  const int n = nt * 16 + r;
  const float bias = bx[n];
#pragma unroll
  for (int v = 0; v < 8; ++v) {
    const int m = mt * 16 + half * 8 + v;
    xg[(size_t)m * G4 + n] = acc[v] + bias;
  }
}

// ---- Stage 4: persistent recurrent scan, one workgroup per direction ------
// 32 waves; wave w owns h-columns [16w,16w+16) and the matching f/i/c~/o gate
// N-tiles {w, w+32, w+64, w+96}. State h (bf16) + c (f32) live in LDS; the
// per-step GEMM is [16,512]x[512,2048] = 64 bf16 WMMAs per wave.
__global__ __launch_bounds__(1024) void k_lstm_scan(
    const float* __restrict__ xgf, const float* __restrict__ xgb,
    const __bf16* __restrict__ wfh, const __bf16* __restrict__ wbh,
    const float* __restrict__ bfh, const float* __restrict__ bbh,
    __bf16* __restrict__ hout /* [T][B][2H] bf16 */) {
  __shared__ __align__(16) __bf16 h_sh[BDIM * HDIM];   // 16 KB
  __shared__ float  c_sh[BDIM * HDIM];                 // 32 KB
  const int dir = blockIdx.x;            // 0 = forward, 1 = backward
  const float*  xg = dir ? xgb : xgf;
  const __bf16* wh = dir ? wbh : wfh;
  const float*  bh = dir ? bbh : bfh;

  const int tid  = threadIdx.x;
  const int wave = tid >> 5;             // 0..31
  const int lane = tid & 31;
  const int r = lane & 15, half = lane >> 4;

  for (int i = tid; i < BDIM * HDIM; i += 1024) {
    h_sh[i] = (__bf16)0.0f;
    c_sh[i] = 0.0f;
  }
  __syncthreads();

  const int j = wave * 16 + r;           // h column this lane updates

  for (int s = 0; s < TDIM; ++s) {
    const int t = dir ? (TDIM - 1 - s) : s;
    v8f acc0 = {}, acc1 = {}, acc2 = {}, acc3 = {};
    for (int k0 = 0; k0 < HDIM; k0 += 32) {
      v16bf a  = load_tile16(h_sh + k0, HDIM, lane);
      v16bf b0 = load_tile16(wh + (size_t)(0 * HDIM + wave * 16) * HDIM + k0, HDIM, lane);
      v16bf b1 = load_tile16(wh + (size_t)(1 * HDIM + wave * 16) * HDIM + k0, HDIM, lane);
      v16bf b2 = load_tile16(wh + (size_t)(2 * HDIM + wave * 16) * HDIM + k0, HDIM, lane);
      v16bf b3 = load_tile16(wh + (size_t)(3 * HDIM + wave * 16) * HDIM + k0, HDIM, lane);
      acc0 = wmma_bf16(a, b0, acc0);
      acc1 = wmma_bf16(a, b1, acc1);
      acc2 = wmma_bf16(a, b2, acc2);
      acc3 = wmma_bf16(a, b3, acc3);
    }
    __syncthreads();                     // all reads of old h done
    const float bf_ = bh[0 * HDIM + j], bi_ = bh[1 * HDIM + j];
    const float bc_ = bh[2 * HDIM + j], bo_ = bh[3 * HDIM + j];
#pragma unroll
    for (int v = 0; v < 8; ++v) {
      const int bb = half * 8 + v;
      const size_t row = (size_t)t * BDIM + bb;
      float fg = sigmoid_(acc0[v] + xg[row * G4 + 0 * HDIM + j] + bf_);
      float ig = sigmoid_(acc1[v] + xg[row * G4 + 1 * HDIM + j] + bi_);
      float cg = tanh_   (acc2[v] + xg[row * G4 + 2 * HDIM + j] + bc_);
      float og = sigmoid_(acc3[v] + xg[row * G4 + 3 * HDIM + j] + bo_);
      float c = fg * c_sh[bb * HDIM + j] + ig * cg;
      c_sh[bb * HDIM + j] = c;
      float h = og * tanh_(c);
      __bf16 hb = (__bf16)h;
      h_sh[bb * HDIM + j] = hb;
      hout[row * K2 + dir * HDIM + j] = hb;   // concat [hf|hb] for the FC
    }
    __syncthreads();                     // new h visible before next GEMM
  }
}

// ---- Stage 5: logits = out @ fc_w^T + fc_b --------------------------------
// Block = 8 waves covers a 128x128 tile of [8192, 32000]. The 8 waves share
// the same 128 weight rows, so B is staged into LDS once per block via
// double-buffered global_load_async_to_lds_b128 (ASYNCcnt), 64-wide K chunks.
// Each wave computes one 16-row strip x 8 N-tiles, reusing A 8x per K step.
#define KCH 64                 // K elements per staged chunk
#define NCHUNK (K2 / KCH)      // 16 chunks

// Stage B chunk kc (128 rows x 64 K bf16 = 16 KB) into LDS at lds_base.
// 256 threads: thread -> (row = tid>>1, 64-byte segment = tid&1), 4x b128.
__device__ __forceinline__ void stage_b_chunk(const __bf16* __restrict__ W,
                                              int ng, int kc,
                                              unsigned int lds_base, int tid) {
  const int row = tid >> 1;
  const int seg = tid & 1;
  const __bf16* g = W + (size_t)(ng * 128 + row) * K2 + kc * KCH + seg * 32;
  const unsigned int l = lds_base + (unsigned int)(row * 128 + seg * 64);
#pragma unroll
  for (int i = 0; i < 4; ++i) async_copy_b128(l + i * 16, g + i * 8);
}

__global__ __launch_bounds__(256) void k_fc_gemm(
    const __bf16* __restrict__ A /* [8192, 1024] rows r=t*16+b */,
    const __bf16* __restrict__ W /* [32000, 1024] bf16 */,
    const float* __restrict__ bias, float* __restrict__ out /* [B,T,V] */) {
  __shared__ __align__(16) __bf16 Bsh[2][128][KCH];   // 2 x 16 KB
  const int tid  = threadIdx.x;
  const int lane = tid & 31;
  const int wave = tid >> 5;
  const int mt = blockIdx.x * 8 + wave;  // 0..511
  const int ng = blockIdx.y;             // 0..249 -> N-tiles ng*8 .. ng*8+7
  const __bf16* abase = A + (size_t)mt * 16 * K2;
  const unsigned int lds0 = (unsigned int)(uintptr_t)&Bsh[0][0][0];
  const unsigned int lds1 = (unsigned int)(uintptr_t)&Bsh[1][0][0];

  stage_b_chunk(W, ng, 0, lds0, tid);    // prefetch chunk 0

  v8f acc[8] = {};
  for (int kc = 0; kc < NCHUNK; ++kc) {
    if (kc + 1 < NCHUNK) {
      // Issue next chunk's DMA into the other buffer (last read at kc-1,
      // protected by the trailing barrier), then wait only for the 4 older
      // in-flight b128s (current chunk) -> DMA overlaps this chunk's math.
      stage_b_chunk(W, ng, kc + 1, ((kc + 1) & 1) ? lds1 : lds0, tid);
      asm volatile("s_wait_asynccnt 0x4" ::: "memory");
    } else {
      asm volatile("s_wait_asynccnt 0x0" ::: "memory");
    }
    __syncthreads();                     // chunk kc visible to all waves
    const __bf16* bb = (kc & 1) ? &Bsh[1][0][0] : &Bsh[0][0][0];
#pragma unroll
    for (int kk = 0; kk < KCH; kk += 32) {
      v16bf a = load_tile16(abase + kc * KCH + kk, K2, lane);
#pragma unroll
      for (int q = 0; q < 8; ++q) {
        v16bf b = load_tile16(bb + q * 16 * KCH + kk, KCH, lane);
        acc[q] = wmma_bf16(a, b, acc[q]);
      }
    }
    __syncthreads();                     // all reads done; buffer reusable
  }

  const int r = lane & 15, half = lane >> 4;
#pragma unroll
  for (int q = 0; q < 8; ++q) {
    const int n = ng * 128 + q * 16 + r;
    const float bv = bias[n];
#pragma unroll
    for (int v = 0; v < 8; ++v) {
      const int m = mt * 16 + half * 8 + v;   // m = t*16 + b
      const int t = m >> 4, bb2 = m & 15;
      out[((size_t)bb2 * TDIM + t) * (size_t)VDIM + n] = acc[q][v] + bv;
    }
  }
}

// ---------------------------------------------------------------------------
extern "C" void kernel_launch(void* const* d_in, const int* in_sizes, int n_in,
                              void* d_out, int out_size, void* d_ws,
                              size_t ws_size, hipStream_t stream) {
  (void)in_sizes; (void)n_in; (void)out_size; (void)ws_size;
  const int*   x    = (const int*)  d_in[0];
  const float* emb  = (const float*)d_in[1];
  const float* Wf_x = (const float*)d_in[2];
  const float* bf_x = (const float*)d_in[3];
  const float* Wf_h = (const float*)d_in[4];
  const float* bf_h = (const float*)d_in[5];
  const float* Wb_x = (const float*)d_in[6];
  const float* bb_x = (const float*)d_in[7];
  const float* Wb_h = (const float*)d_in[8];
  const float* bb_h = (const float*)d_in[9];
  const float* fc_w = (const float*)d_in[10];
  const float* fc_b = (const float*)d_in[11];
  float* out = (float*)d_out;

  // Workspace layout (~233 MB): bf16 activations/weights + f32 xg buffers.
  char* p = (char*)d_ws;
  auto take = [&](size_t bytes) {
    char* q = p;
    p += (bytes + 255) & ~(size_t)255;
    return q;
  };
  __bf16* e_bf   = (__bf16*)take((size_t)ROWS * HDIM * 2);
  __bf16* wfx_bf = (__bf16*)take((size_t)G4 * HDIM * 2);
  __bf16* wbx_bf = (__bf16*)take((size_t)G4 * HDIM * 2);
  __bf16* wfh_bf = (__bf16*)take((size_t)G4 * HDIM * 2);
  __bf16* wbh_bf = (__bf16*)take((size_t)G4 * HDIM * 2);
  __bf16* fcw_bf = (__bf16*)take((size_t)VDIM * K2 * 2);
  float*  xg_f   = (float*) take((size_t)ROWS * G4 * 4);
  float*  xg_b   = (float*) take((size_t)ROWS * G4 * 4);
  __bf16* hout   = (__bf16*)take((size_t)ROWS * K2 * 2);

  // 1) embedding gather -> bf16, time-major rows
  {
    size_t n = (size_t)ROWS * HDIM / 8;
    k_gather_embed<<<(unsigned)((n + 255) / 256), 256, 0, stream>>>(x, emb, e_bf);
  }
  // 2) weight conversions -> bf16
  auto cvt = [&](const float* s, __bf16* d, size_t n) {
    long n8 = (long)(n / 8);
    k_cvt_bf16<<<(unsigned)((n8 + 255) / 256), 256, 0, stream>>>(s, d, n8);
  };
  cvt(Wf_x, wfx_bf, (size_t)G4 * HDIM);
  cvt(Wb_x, wbx_bf, (size_t)G4 * HDIM);
  cvt(Wf_h, wfh_bf, (size_t)G4 * HDIM);
  cvt(Wb_h, wbh_bf, (size_t)G4 * HDIM);
  cvt(fc_w, fcw_bf, (size_t)VDIM * K2);
  // 3) input-side gate GEMMs (both directions via blockIdx.z)
  k_xg_gemm<<<dim3(8192, 1, 2), 256, 0, stream>>>(e_bf, wfx_bf, wbx_bf,
                                                  bf_x, bb_x, xg_f, xg_b);
  // 4) recurrent scans: one persistent WGP per direction
  k_lstm_scan<<<2, 1024, 0, stream>>>(xg_f, xg_b, wfh_bf, wbh_bf,
                                      bf_h, bb_h, hout);
  // 5) vocab projection -> [B, T, V] fp32
  k_fc_gemm<<<dim3(64, 250), 256, 0, stream>>>(hout, fcw_bf, fc_b, out);
}
